// KeyValueSITSelfAttention_66391604461716
// MI455X (gfx1250) — matrix operand order repaired and checked by reading, not compiled, not run on hardware
//
#include <hip/hip_runtime.h>

// ---------------------------------------------------------------------------
// MI455X (gfx1250) fused transformer block, fp32 reference accuracy via
// bf16x3 split-WMMA (hi/lo decomposition, fp32 accumulate).
// Matrix math: v_wmma_f32_16x16x32_bf16.
// Tile staging: global_load_async_to_lds_b128 (ASYNCcnt) + double buffering.
// ---------------------------------------------------------------------------

typedef __bf16 bf16_t;
typedef __attribute__((ext_vector_type(8)))  float  v8f;
typedef __attribute__((ext_vector_type(16))) __bf16 v16bf;
typedef __attribute__((ext_vector_type(8)))  __bf16 v8bf;

#define BB 8
#define TT 2048
#define DD 1024
#define MT (BB * TT)          // 16384 total rows

// --- CDNA5 async global->LDS copy (16 B per lane), ASYNCcnt-tracked --------
__device__ __forceinline__ void async_ld16(const bf16_t* gaddr, const bf16_t* lds) {
  asm volatile("global_load_async_to_lds_b128 %0, %1, off"
               :: "v"((unsigned)(unsigned long long)lds), "v"(gaddr)
               : "memory");
}
__device__ __forceinline__ void wait_async0() {
  asm volatile("s_wait_asynccnt 0x0" ::: "memory");
}

// ---------------------------------------------------------------------------
// fp32 -> (hi, lo) bf16 split
// ---------------------------------------------------------------------------
__global__ __launch_bounds__(256)
void cvt_hilo_kernel(const float* __restrict__ in, bf16_t* __restrict__ h,
                     bf16_t* __restrict__ l, long n) {
  long i = (long)blockIdx.x * 256 + threadIdx.x;
  if (i < n) {
    float f = in[i];
    bf16_t hh = (bf16_t)f;
    h[i] = hh;
    l[i] = (bf16_t)(f - (float)hh);
  }
}

// ---------------------------------------------------------------------------
// bf16x3 WMMA GEMM:  fp32-accurate C = A * op(B), fp32 accumulate.
//   BLAY=0 : B operand is W[n][k] (reduction contiguous) -> full async DB
//   BLAY=1 : B operand is V[k][n] ([K][N] row-major); A async DB, B reg-staged
//            double-buffered LDS transpose.
//   OUTM=0 : store fp32 C.  OUTM=1 : store hi/lo bf16 C.
//   CAUSAL : skip 128x128 blocks strictly above the diagonal (M==N==T).
// Block = 256 thr = 8 waves; block tile 128x128; wave tile 32x64; BLK_K = 32.
// ---------------------------------------------------------------------------
template<int BLAY, int OUTM, bool BIAS, bool RELU, bool CAUSAL>
__global__ __launch_bounds__(256)
void gemm_bf16x3(const bf16_t* __restrict__ Ah, const bf16_t* __restrict__ Al,
                 const bf16_t* __restrict__ Bh, const bf16_t* __restrict__ Bl,
                 const float* __restrict__ bias,
                 float* __restrict__ Cf, bf16_t* __restrict__ Ch,
                 bf16_t* __restrict__ Cl,
                 int M, int N, int K, int lda, int ldb, int ldc,
                 long sA, long sB, long sC) {
  if (CAUSAL && blockIdx.y > blockIdx.x) return;  // above-diagonal: no work

  constexpr int LDT = 40;            // padded LDS row stride (80 B, bank-safe)
  constexpr int TILE = 128 * LDT;    // bf16 elems per tile buffer
  __shared__ bf16_t As_h[2 * TILE], As_l[2 * TILE];
  __shared__ bf16_t Bs_h[2 * TILE], Bs_l[2 * TILE];

  const int tid = threadIdx.x;
  const long bz = blockIdx.z;
  Ah += bz * sA;  Al += bz * sA;
  Bh += bz * sB;  Bl += bz * sB;
  const long cbase = bz * sC;

  const int m0 = blockIdx.x * 128;
  const int n0 = blockIdx.y * 128;

  const int wave = tid >> 5;
  const int lane = tid & 31;
  const int lh   = lane & 15;
  const int hi16 = lane >> 4;          // half-wave selects K offset
  const int wm   = (wave & 3) * 32;    // wave M offset in block tile
  const int wn   = (wave >> 2) * 64;   // wave N offset in block tile

  v8f acc[2][4];
#pragma unroll
  for (int i = 0; i < 2; ++i)
#pragma unroll
    for (int j = 0; j < 4; ++j)
#pragma unroll
      for (int e = 0; e < 8; ++e) acc[i][j][e] = 0.f;

  // fragment loader (ISA 16-bit A 16x32 / B 32x16 wave32 layouts)
  auto frags = [&](int buf, v16bf afh[2], v16bf afl[2], v16bf bfh[4],
                   v16bf bfl[4]) {
    const int ob = buf * TILE;
#pragma unroll
    for (int i = 0; i < 2; ++i) {
      int row = wm + i * 16 + lh;
      int kb = hi16 * 8;   // lanes 16-31: K+8 (vgpr0-3) / K+24 (vgpr4-7)
      v8bf x0 = *(const v8bf*)&As_h[ob + row * LDT + kb];
      v8bf x1 = *(const v8bf*)&As_h[ob + row * LDT + 16 + kb];
      afh[i] = __builtin_shufflevector(x0, x1, 0,1,2,3,4,5,6,7,8,9,10,11,12,13,14,15);
      v8bf y0 = *(const v8bf*)&As_l[ob + row * LDT + kb];
      v8bf y1 = *(const v8bf*)&As_l[ob + row * LDT + 16 + kb];
      afl[i] = __builtin_shufflevector(y0, y1, 0,1,2,3,4,5,6,7,8,9,10,11,12,13,14,15);
    }
#pragma unroll
    for (int j = 0; j < 4; ++j) {
      int n = wn + j * 16 + lh;
      int kb = hi16 * 8;   // lanes 16-31: K+8 (vgpr0-3) / K+24 (vgpr4-7)
      v8bf x0 = *(const v8bf*)&Bs_h[ob + n * LDT + kb];
      v8bf x1 = *(const v8bf*)&Bs_h[ob + n * LDT + 16 + kb];
      bfh[j] = __builtin_shufflevector(x0, x1, 0,1,2,3,4,5,6,7,8,9,10,11,12,13,14,15);
      v8bf y0 = *(const v8bf*)&Bs_l[ob + n * LDT + kb];
      v8bf y1 = *(const v8bf*)&Bs_l[ob + n * LDT + 16 + kb];
      bfl[j] = __builtin_shufflevector(y0, y1, 0,1,2,3,4,5,6,7,8,9,10,11,12,13,14,15);
    }
  };
  // bf16x3 accumulation: Ah*Bh + Ah*Bl + Al*Bh (fp32 accumulate)
  auto mma = [&](v16bf afh[2], v16bf afl[2], v16bf bfh[4], v16bf bfl[4]) {
#pragma unroll
    for (int i = 0; i < 2; ++i)
#pragma unroll
      for (int j = 0; j < 4; ++j) {
        acc[i][j] = __builtin_amdgcn_wmma_f32_16x16x32_bf16(
            false, afl[i], false, bfh[j], (short)0, acc[i][j], false, false);
        acc[i][j] = __builtin_amdgcn_wmma_f32_16x16x32_bf16(
            false, afh[i], false, bfl[j], (short)0, acc[i][j], false, false);
        acc[i][j] = __builtin_amdgcn_wmma_f32_16x16x32_bf16(
            false, afh[i], false, bfh[j], (short)0, acc[i][j], false, false);
      }
  };
  // async copy of the A tile (rows k-contiguous) into buffer `buf`
  auto issue_A = [&](int kt, int buf) {
    const int ob = buf * TILE;
#pragma unroll
    for (int p = 0; p < 2; ++p) {
      int idx = p * 256 + tid;         // 512 chunks of 8 bf16 (16 B)
      int r = idx >> 2, c = (idx & 3) * 8;
      long go = (long)(m0 + r) * lda + kt + c;
      async_ld16(Ah + go, &As_h[ob + r * LDT + c]);
      async_ld16(Al + go, &As_l[ob + r * LDT + c]);
    }
  };

  if (BLAY == 0) {
    // ---- both operands k-contiguous: full async double buffering ----
    auto issue_B = [&](int kt, int buf) {
      const int ob = buf * TILE;
#pragma unroll
      for (int p = 0; p < 2; ++p) {
        int idx = p * 256 + tid;
        int r = idx >> 2, c = (idx & 3) * 8;
        long go = (long)(n0 + r) * ldb + kt + c;
        async_ld16(Bh + go, &Bs_h[ob + r * LDT + c]);
        async_ld16(Bl + go, &Bs_l[ob + r * LDT + c]);
      }
    };
    issue_A(0, 0);
    issue_B(0, 0);
    wait_async0();
    __syncthreads();
    int buf = 0;
    for (int kt = 0; kt < K; kt += 32) {
      if (kt + 32 < K) {               // overlap next tile with compute
        issue_A(kt + 32, buf ^ 1);
        issue_B(kt + 32, buf ^ 1);
      }
      v16bf afh[2], afl[2], bfh[4], bfl[4];
      frags(buf, afh, afl, bfh, bfl);
      mma(afh, afl, bfh, bfl);
      wait_async0();      // next tile landed in LDS (this wave's copies)
      __syncthreads();    // all waves: done reading buf, next buf visible
      buf ^= 1;
    }
  } else {
    // ---- B = V[k][n]: A async DB; B reg-staged + LDS-transposed DB ----
    auto load_B_regs = [&](int kt, uint4 vh[2], uint4 vl[2]) {
#pragma unroll
      for (int p = 0; p < 2; ++p) {
        int idx = p * 256 + tid;       // 32 k-rows x 16 chunks of 8 n
        int k = idx >> 4, c = (idx & 15) * 8;
        long go = (long)(kt + k) * ldb + n0 + c;
        vh[p] = *(const uint4*)(Bh + go);
        vl[p] = *(const uint4*)(Bl + go);
      }
    };
    auto store_B_lds = [&](int buf, const uint4 vh[2], const uint4 vl[2]) {
      const int ob = buf * TILE;
#pragma unroll
      for (int p = 0; p < 2; ++p) {
        int idx = p * 256 + tid;
        int k = idx >> 4, c = (idx & 15) * 8;
        const bf16_t* eh = (const bf16_t*)&vh[p];
        const bf16_t* el = (const bf16_t*)&vl[p];
#pragma unroll
        for (int j = 0; j < 8; ++j) {
          Bs_h[ob + (c + j) * LDT + k] = eh[j];
          Bs_l[ob + (c + j) * LDT + k] = el[j];
        }
      }
    };
    issue_A(0, 0);
    {
      uint4 vh[2], vl[2];
      load_B_regs(0, vh, vl);
      store_B_lds(0, vh, vl);
    }
    wait_async0();
    __syncthreads();
    int buf = 0;
    for (int kt = 0; kt < K; kt += 32) {
      const bool next = (kt + 32 < K);
      uint4 vh[2], vl[2];
      if (next) {
        issue_A(kt + 32, buf ^ 1);     // async: overlaps with compute
        load_B_regs(kt + 32, vh, vl);  // global loads hide under compute
      }
      v16bf afh[2], afl[2], bfh[4], bfl[4];
      frags(buf, afh, afl, bfh, bfl);
      mma(afh, afl, bfh, bfl);
      if (next) store_B_lds(buf ^ 1, vh, vl);  // transpose into other buffer
      wait_async0();
      __syncthreads();                 // fences DScnt + publishes both tiles
      buf ^= 1;
    }
  }

  // ---- epilogue (C layout: vgpr e -> M = e + 8*hi16, N = lane&15) ----
#pragma unroll
  for (int i = 0; i < 2; ++i)
#pragma unroll
    for (int j = 0; j < 4; ++j) {
      int col  = n0 + wn + j * 16 + lh;
      int row0 = m0 + wm + i * 16 + hi16 * 8;
      float bv = BIAS ? bias[col] : 0.f;
#pragma unroll
      for (int e = 0; e < 8; ++e) {
        float v = acc[i][j][e] + bv;
        if (RELU) v = v > 0.f ? v : 0.f;
        long o = cbase + (long)(row0 + e) * ldc + col;
        if (OUTM == 0) {
          Cf[o] = v;
        } else {
          bf16_t hh = (bf16_t)v;
          Ch[o] = hh;
          Cl[o] = (bf16_t)(v - (float)hh);
        }
      }
    }
}

// ---------------------------------------------------------------------------
// Row softmax + causal bound + post-softmax SIT mask.
// One wave32 per row. Denominator sums ALL causal keys (mask only zeroes
// numerator terms, matching reference semantics). Emits P as hi/lo bf16.
// ---------------------------------------------------------------------------
__global__ __launch_bounds__(256)
void softmax_mask_kernel(const float* __restrict__ scores,
                         const float* __restrict__ sit,
                         bf16_t* __restrict__ Ph, bf16_t* __restrict__ Pl) {
  const int wave = threadIdx.x >> 5, lane = threadIdx.x & 31;
  const int r = blockIdx.x * 8 + wave;      // [0, B*T)
  const int b = r >> 11, t = r & (TT - 1);
  const float sc = 0.03125f;                // 1/sqrt(1024)
  const float* srow = scores + (long)r * TT;

  float m = -3.4e38f;
  for (int s = lane; s <= t; s += 32) m = fmaxf(m, srow[s] * sc);
  for (int o = 16; o; o >>= 1) m = fmaxf(m, __shfl_xor(m, o, 32));

  float l = 0.f;
  for (int s = lane; s <= t; s += 32) l += __expf(srow[s] * sc - m);
  for (int o = 16; o; o >>= 1) l += __shfl_xor(l, o, 32);
  const float inv = 1.f / l;

  const float qm = sit[b * TT + t];
  for (int s = lane; s < TT; s += 32) {
    float p = 0.f;
    if (s <= t && qm != 0.f && sit[b * TT + s] != 0.f)
      p = __expf(srow[s] * sc - m) * inv;
    bf16_t hh = (bf16_t)p;
    long o = (long)r * TT + s;
    Ph[o] = hh;
    Pl[o] = (bf16_t)(p - (float)hh);
  }
}

// ---------------------------------------------------------------------------
// out = LayerNorm(a + b) * g + beta.  One wave32 per 1024-wide row; row held
// in registers between the reduction and the normalize/store pass.
// STOREHL additionally emits hi/lo bf16 (feeds the next GEMM directly).
// ---------------------------------------------------------------------------
template<bool STOREHL>
__global__ __launch_bounds__(256)
void add_ln_kernel(const float* __restrict__ a, const float* __restrict__ b,
                   const float* __restrict__ g, const float* __restrict__ beta,
                   float* __restrict__ out, bf16_t* __restrict__ oh,
                   bf16_t* __restrict__ ol) {
  const int wave = threadIdx.x >> 5, lane = threadIdx.x & 31;
  const long row = (long)blockIdx.x * 8 + wave;
  const float4* a4 = (const float4*)(a + row * DD);
  const float4* b4 = (const float4*)(b + row * DD);
  float4 y[8];
  float s1 = 0.f, s2 = 0.f;
#pragma unroll
  for (int it = 0; it < 8; ++it) {
    int i = it * 32 + lane;
    float4 av = a4[i], bv = b4[i];
    float4 yv = {av.x + bv.x, av.y + bv.y, av.z + bv.z, av.w + bv.w};
    y[it] = yv;
    s1 += yv.x + yv.y + yv.z + yv.w;
    s2 += yv.x * yv.x + yv.y * yv.y + yv.z * yv.z + yv.w * yv.w;
  }
  for (int o = 16; o; o >>= 1) {
    s1 += __shfl_xor(s1, o, 32);
    s2 += __shfl_xor(s2, o, 32);
  }
  const float mu = s1 * (1.f / DD);
  const float var = s2 * (1.f / DD) - mu * mu;
  const float rs = rsqrtf(var + 1e-5f);
#pragma unroll
  for (int it = 0; it < 8; ++it) {
    int i = it * 32 + lane;
    float4 gv = ((const float4*)g)[i];
    float4 bt = ((const float4*)beta)[i];
    float4 yv = y[it];
    float4 ov = {(yv.x - mu) * rs * gv.x + bt.x,
                 (yv.y - mu) * rs * gv.y + bt.y,
                 (yv.z - mu) * rs * gv.z + bt.z,
                 (yv.w - mu) * rs * gv.w + bt.w};
    ((float4*)(out + row * DD))[i] = ov;
    if (STOREHL) {
      long base = row * DD + (long)i * 4;
      float vals[4] = {ov.x, ov.y, ov.z, ov.w};
#pragma unroll
      for (int e = 0; e < 4; ++e) {
        bf16_t hh = (bf16_t)vals[e];
        oh[base + e] = hh;
        ol[base + e] = (bf16_t)(vals[e] - (float)hh);
      }
    }
  }
}

// ---------------------------------------------------------------------------
// Host orchestration.  Workspace layout (~660 MB, with deliberate reuse).
// ---------------------------------------------------------------------------
extern "C" void kernel_launch(void* const* d_in, const int* in_sizes, int n_in,
                              void* d_out, int out_size, void* d_ws,
                              size_t ws_size, hipStream_t stream) {
  const float* x   = (const float*)d_in[0];
  const float* sit = (const float*)d_in[1];
  const float* Wq  = (const float*)d_in[2];
  const float* Wk  = (const float*)d_in[3];
  const float* Wv  = (const float*)d_in[4];
  const float* g1  = (const float*)d_in[5];
  const float* be1 = (const float*)d_in[6];
  const float* Wf1 = (const float*)d_in[7];
  const float* bia1 = (const float*)d_in[8];
  const float* Wf2 = (const float*)d_in[9];
  const float* bia2 = (const float*)d_in[10];
  const float* g2  = (const float*)d_in[11];
  const float* be2 = (const float*)d_in[12];
  float* out = (float*)d_out;

  char* w = (char*)d_ws;
  size_t off = 0;
  auto alloc = [&](size_t bytes) -> char* {
    char* p = w + off;
    off = (off + bytes + 255) & ~(size_t)255;
    return p;
  };
  const size_t xe = (size_t)MT * DD;        // 16M elements
  const size_t we = (size_t)DD * DD;        // 1M elements
  const size_t se = (size_t)BB * TT * TT;   // 32M elements

  bf16_t* x_h = (bf16_t*)alloc(xe * 2);
  bf16_t* x_l = (bf16_t*)alloc(xe * 2);
  bf16_t* Wq_h = (bf16_t*)alloc(we * 2); bf16_t* Wq_l = (bf16_t*)alloc(we * 2);
  bf16_t* Wk_h = (bf16_t*)alloc(we * 2); bf16_t* Wk_l = (bf16_t*)alloc(we * 2);
  bf16_t* Wv_h = (bf16_t*)alloc(we * 2); bf16_t* Wv_l = (bf16_t*)alloc(we * 2);
  bf16_t* W1_h = (bf16_t*)alloc(we * 2); bf16_t* W1_l = (bf16_t*)alloc(we * 2);
  bf16_t* W2_h = (bf16_t*)alloc(we * 2); bf16_t* W2_l = (bf16_t*)alloc(we * 2);
  bf16_t* q_h = (bf16_t*)alloc(xe * 2); bf16_t* q_l = (bf16_t*)alloc(xe * 2);
  bf16_t* k_h = (bf16_t*)alloc(xe * 2); bf16_t* k_l = (bf16_t*)alloc(xe * 2);
  bf16_t* v_h = (bf16_t*)alloc(xe * 2); bf16_t* v_l = (bf16_t*)alloc(xe * 2);
  float* scores = (float*)alloc(se * 4);
  bf16_t* P_h = (bf16_t*)alloc(se * 2);
  bf16_t* P_l = (bf16_t*)alloc(se * 2);
  float* attn = (float*)alloc(xe * 4);
  float* hbuf = (float*)alloc(xe * 4);
  // dead-buffer reuse:
  bf16_t* h_h = x_h;   bf16_t* h_l = x_l;      // x_hl dead after QKV
  bf16_t* t1_h = q_h;  bf16_t* t1_l = q_l;     // q dead after scores
  float*  ff   = scores;                        // scores dead after softmax

  // 1) split fp32 -> bf16 hi/lo
  cvt_hilo_kernel<<<(int)((xe + 255) / 256), 256, 0, stream>>>(x, x_h, x_l, (long)xe);
  cvt_hilo_kernel<<<(int)((we + 255) / 256), 256, 0, stream>>>(Wq, Wq_h, Wq_l, (long)we);
  cvt_hilo_kernel<<<(int)((we + 255) / 256), 256, 0, stream>>>(Wk, Wk_h, Wk_l, (long)we);
  cvt_hilo_kernel<<<(int)((we + 255) / 256), 256, 0, stream>>>(Wv, Wv_h, Wv_l, (long)we);
  cvt_hilo_kernel<<<(int)((we + 255) / 256), 256, 0, stream>>>(Wf1, W1_h, W1_l, (long)we);
  cvt_hilo_kernel<<<(int)((we + 255) / 256), 256, 0, stream>>>(Wf2, W2_h, W2_l, (long)we);

  // 2) Q, K, V projections (store hi/lo bf16 directly)
  dim3 gQKV(MT / 128, DD / 128, 1);
  gemm_bf16x3<0, 1, false, false, false><<<gQKV, 256, 0, stream>>>(
      x_h, x_l, Wq_h, Wq_l, nullptr, nullptr, q_h, q_l,
      MT, DD, DD, DD, DD, DD, 0, 0, 0);
  gemm_bf16x3<0, 1, false, false, false><<<gQKV, 256, 0, stream>>>(
      x_h, x_l, Wk_h, Wk_l, nullptr, nullptr, k_h, k_l,
      MT, DD, DD, DD, DD, DD, 0, 0, 0);
  gemm_bf16x3<0, 1, false, false, false><<<gQKV, 256, 0, stream>>>(
      x_h, x_l, Wv_h, Wv_l, nullptr, nullptr, v_h, v_l,
      MT, DD, DD, DD, DD, DD, 0, 0, 0);

  // 3) scores = Q K^T (batched, causal block-skip); softmax applies 1/32
  dim3 gSc(TT / 128, TT / 128, BB);
  gemm_bf16x3<0, 0, false, false, true><<<gSc, 256, 0, stream>>>(
      q_h, q_l, k_h, k_l, nullptr, scores, nullptr, nullptr,
      TT, TT, DD, DD, DD, TT, (long)TT * DD, (long)TT * DD, (long)TT * TT);

  // 4) softmax + causal + post-softmax SIT mask -> P (hi/lo bf16)
  softmax_mask_kernel<<<MT / 8, 256, 0, stream>>>(scores, sit, P_h, P_l);

  // 5) attn = P V (batched; B operand is [K][N] row-major)
  dim3 gPV(TT / 128, DD / 128, BB);
  gemm_bf16x3<1, 0, false, false, false><<<gPV, 256, 0, stream>>>(
      P_h, P_l, v_h, v_l, nullptr, attn, nullptr, nullptr,
      TT, DD, TT, TT, DD, DD, (long)TT * TT, (long)TT * DD, (long)TT * DD);

  // 6) h = LN(attn + x)  (also emit hi/lo bf16 for the FFN)
  add_ln_kernel<true><<<MT / 8, 256, 0, stream>>>(attn, x, g1, be1, hbuf, h_h, h_l);

  // 7) t1 = relu(h W1^T + b1)  (hi/lo bf16 out)
  gemm_bf16x3<0, 1, true, true, false><<<gQKV, 256, 0, stream>>>(
      h_h, h_l, W1_h, W1_l, bia1, nullptr, t1_h, t1_l,
      MT, DD, DD, DD, DD, DD, 0, 0, 0);

  // 8) ff = t1 W2^T + b2
  gemm_bf16x3<0, 0, true, false, false><<<gQKV, 256, 0, stream>>>(
      t1_h, t1_l, W2_h, W2_l, bia2, ff, nullptr, nullptr,
      MT, DD, DD, DD, DD, DD, 0, 0, 0);

  // 9) out = LN(ff + h)
  add_ln_kernel<false><<<MT / 8, 256, 0, stream>>>(ff, hbuf, g2, be2, out,
                                                   nullptr, nullptr);
}
